// TGATModel_70806830841993
// MI455X (gfx1250) — matrix operand order repaired and checked by reading, not compiled, not run on hardware
//
#include <hip/hip_runtime.h>
#include <hip/hip_bf16.h>
#include <math.h>

#define TD  32
#define HID 64

typedef __attribute__((ext_vector_type(16))) __bf16 bf16x16;
typedef __attribute__((ext_vector_type(8)))  float  f32x8;

union ABu { bf16x16 v; unsigned short u[16]; uint4 q[2]; };
union DFu { f32x8 v; float f[8]; };

// float -> bf16 bits, round-to-nearest-even (ignore NaN corner)
__device__ __forceinline__ unsigned short f2bf(float f) {
    unsigned int x = __float_as_uint(f);
    unsigned int r = x + 0x7FFFu + ((x >> 16) & 1u);
    return (unsigned short)(r >> 16);
}
// order-preserving float->int key for atomicMax-based segment max
__device__ __forceinline__ int fkey(float f) {
    int i = __float_as_int(f);
    return (i >= 0) ? i : (i ^ 0x7FFFFFFF);
}
__device__ __forceinline__ float funkey(int k) {
    return __int_as_float((k >= 0) ? k : (k ^ 0x7FFFFFFF));
}

// ---- WMMA fragment loaders (layouts per CDNA5 ISA §7.12.2) ----
// A (16x32 bf16): lane L holds row L&15; half=L>>4 selects K block.
// elements e<8 -> k = kc + half*8 + e ; e>=8 -> k = kc + 16 + half*8 + (e-8)
__device__ __forceinline__ bf16x16 load_a_frag(const unsigned short* A, int row,
                                               int strideK, int kc, int half) {
    ABu a;
    const unsigned short* p = A + row * strideK + kc + half * 8;
    a.q[0] = *(const uint4*)(p);
    a.q[1] = *(const uint4*)(p + 16);
    return a.v;
}
// B stored pre-swizzled: Bsw[((chunk*HID + col)*2 + half)*16 + e]
__device__ __forceinline__ bf16x16 load_b_frag(const unsigned short* Bsw, int chunk,
                                               int col, int half) {
    ABu b;
    const unsigned short* p = Bsw + (((chunk * HID + col) * 2 + half) << 4);
    b.q[0] = *(const uint4*)(p);
    b.q[1] = *(const uint4*)(p + 8);
    return b.v;
}

// D(16x16 f32) tile = A(16xKA, LDS row-major) x Bsw(KAx64 swizzled), cols [colBase,colBase+16)
template<int KA>
__device__ __forceinline__ f32x8 wmma_rm(const unsigned short* A, int rowBase,
                                         const unsigned short* Bsw, int colBase,
                                         f32x8 acc) {
    const int lane = threadIdx.x & 31;
    const int half = lane >> 4;
    const int r16  = lane & 15;
#pragma unroll
    for (int kc = 0; kc < KA; kc += 32) {
        bf16x16 a = load_a_frag(A, rowBase + r16, KA, kc, half);
        bf16x16 b = load_b_frag(Bsw, kc >> 5, colBase + r16, half);
        acc = __builtin_amdgcn_wmma_f32_16x16x32_bf16(false, a, false, b,
                                                      (short)0, acc, false, false);
    }
    return acc;
}

// ---------------- setup kernels ----------------
__global__ void convert_w_swizzle(const float* __restrict__ src, unsigned short* __restrict__ dst,
                                  int Ksrc, int Kpad) {
    int i = blockIdx.x * blockDim.x + threadIdx.x;
    if (i >= Kpad * HID) return;
    int e = i & 15, half = (i >> 4) & 1, col = (i >> 5) & 63, chunk = i >> 11;
    int k = chunk * 32 + ((e < 8) ? 0 : 16) + half * 8 + (e & 7);
    dst[i] = (k < Ksrc) ? f2bf(src[k * HID + col]) : (unsigned short)0;
}

__global__ void build_h0(const int* __restrict__ u, const int* __restrict__ v,
                         float* __restrict__ h0, int N) {
    int i = blockIdx.x * blockDim.x + threadIdx.x;
    if (i < N) { h0[2 * i] = (float)u[i]; h0[2 * i + 1] = (float)v[i]; }
}

__global__ void init_layer_ws(int* __restrict__ segmax, float* __restrict__ den,
                              float* __restrict__ agg, int N) {
    int i = blockIdx.x * blockDim.x + threadIdx.x;
    if (i < N * HID) agg[i] = 0.0f;
    if (i < N) { segmax[i] = (int)0x80000000; den[i] = 0.0f; }
}

__global__ void zero_vec(float* __restrict__ p, int n) {
    int i = blockIdx.x * blockDim.x + threadIdx.x;
    if (i < n) p[i] = 0.0f;
}

// ---------------- pass 1: logits + segment max ----------------
template<int INF, int KMSG, int KQ>
__global__ __launch_bounds__(256)
void tgat_logits(const float* __restrict__ hin, const float* __restrict__ dt,
                 const int* __restrict__ src, const int* __restrict__ dst,
                 const float* __restrict__ tw, const float* __restrict__ tb,
                 const unsigned short* __restrict__ Wv, const unsigned short* __restrict__ Wk,
                 const unsigned short* __restrict__ Wq,
                 float* __restrict__ logits, int* __restrict__ segmax, int E) {
    __shared__ alignas(16) unsigned short sWv[KMSG * HID];
    __shared__ alignas(16) unsigned short sWk[HID * HID];
    __shared__ alignas(16) unsigned short sWq[KQ * HID];
    __shared__ alignas(16) unsigned short sA [32 * KMSG];
    __shared__ alignas(16) unsigned short sHd[32 * KQ];
    __shared__ alignas(16) unsigned short sMsg[32 * HID];
    __shared__ float sLogit[32];
    __shared__ float sTw[TD], sTb[TD];

    const int tid = threadIdx.x;
    const int base = blockIdx.x * 32;

    for (int i = tid; i < KMSG * HID; i += 256) sWv[i] = Wv[i];
    for (int i = tid; i < HID * HID; i += 256)  sWk[i] = Wk[i];
    for (int i = tid; i < KQ * HID; i += 256)   sWq[i] = Wq[i];
    if (tid < TD) { sTw[tid] = tw[tid]; sTb[tid] = tb[tid]; }
    if (tid < 32) sLogit[tid] = 0.0f;
    __syncthreads();

    for (int i = tid; i < 32 * KMSG; i += 256) {
        int e = i / KMSG, k = i - e * KMSG;
        int g = base + e;
        float v = 0.0f;
        if (g < E) {
            if (k < INF)           v = hin[src[g] * INF + k];
            else if (k < INF + TD) v = cosf(dt[g] * sTw[k - INF] + sTb[k - INF]);
        }
        sA[i] = f2bf(v);
    }
    for (int i = tid; i < 32 * KQ; i += 256) {
        int e = i / KQ, k = i - e * KQ;
        int g = base + e;
        float v = (g < E && k < INF) ? hin[dst[g] * INF + k] : 0.0f;
        sHd[i] = f2bf(v);
    }
    __syncthreads();

    const int wave = tid >> 5;
    const int rowBase = (wave >> 2) * 16;
    const int colBase = (wave & 3) * 16;
    const int lane = tid & 31;
    const int half = lane >> 4;
    const int lcol = lane & 15;
    const f32x8 z = {0.f, 0.f, 0.f, 0.f, 0.f, 0.f, 0.f, 0.f};

    DFu msg; msg.v = wmma_rm<KMSG>(sA, rowBase, sWv, colBase, z);
#pragma unroll
    for (int r = 0; r < 8; ++r) {
        int row = rowBase + r + 8 * half;
        sMsg[row * HID + colBase + lcol] = f2bf(msg.f[r]);
    }
    __syncthreads();

    DFu key; key.v = wmma_rm<HID>(sMsg, rowBase, sWk, colBase, z);
    DFu q;   q.v   = wmma_rm<KQ>(sHd, rowBase, sWq, colBase, z);

#pragma unroll
    for (int r = 0; r < 8; ++r) {
        int row = rowBase + r + 8 * half;
        atomicAdd(&sLogit[row], q.f[r] * key.f[r]);
    }
    __syncthreads();

    if (tid < 32) {
        int g = base + tid;
        if (g < E) {
            float lg = sLogit[tid] * 0.125f;   // 1/sqrt(HID)
            logits[g] = lg;
            atomicMax(&segmax[dst[g]], fkey(lg));
        }
    }
}

// ---------------- pass 2: exp + segment sum ----------------
__global__ void tgat_expsum(float* __restrict__ logits, const int* __restrict__ dst,
                            const int* __restrict__ segmax, float* __restrict__ den, int E) {
    int i = blockIdx.x * blockDim.x + threadIdx.x;
    if (i >= E) return;
    int d = dst[i];
    float e = __expf(logits[i] - funkey(segmax[d]));
    logits[i] = e;                       // in-place: logits buffer now holds ex
    atomicAdd(&den[d], e);
}

// ---------------- pass 3: recompute msg, scatter attn*msg ----------------
template<int INF, int KMSG>
__global__ __launch_bounds__(256)
void tgat_aggregate(const float* __restrict__ hin, const float* __restrict__ dt,
                    const int* __restrict__ src, const int* __restrict__ dst,
                    const float* __restrict__ tw, const float* __restrict__ tb,
                    const unsigned short* __restrict__ Wv,
                    const float* __restrict__ ex, const float* __restrict__ den,
                    float* __restrict__ agg, int E) {
    __shared__ alignas(16) unsigned short sWv[KMSG * HID];
    __shared__ alignas(16) unsigned short sA [32 * KMSG];
    __shared__ float sAttn[32];
    __shared__ int   sDst[32];
    __shared__ float sTw[TD], sTb[TD];

    const int tid = threadIdx.x;
    const int base = blockIdx.x * 32;

    for (int i = tid; i < KMSG * HID; i += 256) sWv[i] = Wv[i];
    if (tid < TD) { sTw[tid] = tw[tid]; sTb[tid] = tb[tid]; }
    if (tid < 32) {
        int g = base + tid;
        int d = (g < E) ? dst[g] : 0;
        sDst[tid] = d;
        sAttn[tid] = (g < E) ? (ex[g] / den[d]) : 0.0f;
    }
    __syncthreads();

    for (int i = tid; i < 32 * KMSG; i += 256) {
        int e = i / KMSG, k = i - e * KMSG;
        int g = base + e;
        float v = 0.0f;
        if (g < E) {
            if (k < INF)           v = hin[src[g] * INF + k];
            else if (k < INF + TD) v = cosf(dt[g] * sTw[k - INF] + sTb[k - INF]);
        }
        sA[i] = f2bf(v);
    }
    __syncthreads();

    const int wave = tid >> 5;
    const int rowBase = (wave >> 2) * 16;
    const int colBase = (wave & 3) * 16;
    const int lane = tid & 31;
    const int half = lane >> 4;
    const int lcol = lane & 15;
    const f32x8 z = {0.f, 0.f, 0.f, 0.f, 0.f, 0.f, 0.f, 0.f};

    DFu msg; msg.v = wmma_rm<KMSG>(sA, rowBase, sWv, colBase, z);

#pragma unroll
    for (int r = 0; r < 8; ++r) {
        int row = rowBase + r + 8 * half;
        int g = base + row;
        if (g < E) {
            float v = msg.f[r] * sAttn[row];
            atomicAdd(&agg[sDst[row] * HID + colBase + lcol], v);
        }
    }
}

// ---------------- pass 4: node update GEMMs + ReLU ----------------
template<int INF, int KIN>
__global__ __launch_bounds__(256)
void tgat_update(const float* __restrict__ hin, const float* __restrict__ agg,
                 const unsigned short* __restrict__ Wo, const unsigned short* __restrict__ Ws,
                 const float* __restrict__ bo, const float* __restrict__ bs,
                 float* __restrict__ hout, int N) {
    __shared__ alignas(16) unsigned short sWo[HID * HID];
    __shared__ alignas(16) unsigned short sWs[KIN * HID];
    __shared__ alignas(16) unsigned short sAgg[32 * HID];
    __shared__ alignas(16) unsigned short sH [32 * KIN];
    __shared__ float sB[HID];

    const int tid = threadIdx.x;
    const int base = blockIdx.x * 32;

    for (int i = tid; i < HID * HID; i += 256) sWo[i] = Wo[i];
    for (int i = tid; i < KIN * HID; i += 256) sWs[i] = Ws[i];
    if (tid < HID) sB[tid] = bo[tid] + bs[tid];
    for (int i = tid; i < 32 * HID; i += 256) {
        int nrow = base + (i >> 6);
        sAgg[i] = (nrow < N) ? f2bf(agg[nrow * HID + (i & 63)]) : (unsigned short)0;
    }
    for (int i = tid; i < 32 * KIN; i += 256) {
        int e = i / KIN, k = i - e * KIN;
        int nrow = base + e;
        float v = (nrow < N && k < INF) ? hin[nrow * INF + k] : 0.0f;
        sH[i] = f2bf(v);
    }
    __syncthreads();

    const int wave = tid >> 5;
    const int rowBase = (wave >> 2) * 16;
    const int colBase = (wave & 3) * 16;
    const int lane = tid & 31;
    const int half = lane >> 4;
    const int lcol = lane & 15;
    const f32x8 z = {0.f, 0.f, 0.f, 0.f, 0.f, 0.f, 0.f, 0.f};

    DFu o; o.v = wmma_rm<HID>(sAgg, rowBase, sWo, colBase, z);
    DFu s; s.v = wmma_rm<KIN>(sH,  rowBase, sWs, colBase, z);

#pragma unroll
    for (int r = 0; r < 8; ++r) {
        int row = rowBase + r + 8 * half;
        int node = base + row;
        int col = colBase + lcol;
        if (node < N) {
            float v = o.f[r] + s.f[r] + sB[col];
            hout[node * HID + col] = fmaxf(v, 0.0f);
        }
    }
}

// ---------------- readout ----------------
__global__ __launch_bounds__(256)
void reduce_sum(const float* __restrict__ h, float* __restrict__ hsum, int N) {
    __shared__ float s[4][HID];
    int c = threadIdx.x & 63, g = threadIdx.x >> 6;
    float acc = 0.0f;
    for (int row = blockIdx.x * 4 + g; row < N; row += gridDim.x * 4)
        acc += h[row * HID + c];
    s[g][c] = acc;
    __syncthreads();
    if (g == 0) atomicAdd(&hsum[c], s[0][c] + s[1][c] + s[2][c] + s[3][c]);
}

__global__ void scorer(const float* __restrict__ hsum,
                       const float* __restrict__ W1, const float* __restrict__ b1,
                       const float* __restrict__ W2, const float* __restrict__ b2,
                       float* __restrict__ out, int N) {
    __shared__ float hg[HID];
    __shared__ float s1[HID];
    int t = threadIdx.x;  // 64 threads
    hg[t] = hsum[t] / (float)N;
    __syncthreads();
    float a = b1[t];
    for (int c = 0; c < HID; ++c) a += hg[c] * W1[c * HID + t];
    s1[t] = fmaxf(a, 0.0f) * W2[t];
    __syncthreads();
    if (t == 0) {
        float sc = b2[0];
        for (int j = 0; j < HID; ++j) sc += s1[j];
        out[0] = sc;
    }
}

// ---------------- host launcher ----------------
extern "C" void kernel_launch(void* const* d_in, const int* in_sizes, int n_in,
                              void* d_out, int out_size, void* d_ws, size_t ws_size,
                              hipStream_t stream) {
    (void)n_in; (void)out_size; (void)ws_size;
    const float* dt  = (const float*)d_in[0];
    const int*   src = (const int*)d_in[1];
    const int*   dst = (const int*)d_in[2];
    const int*   uf  = (const int*)d_in[3];
    const int*   vf  = (const int*)d_in[4];
    const float* tw  = (const float*)d_in[6];
    const float* tb  = (const float*)d_in[7];
    const float* l0_Wv = (const float*)d_in[8];
    const float* l0_Wk = (const float*)d_in[9];
    const float* l0_Wq = (const float*)d_in[10];
    const float* l0_Wo = (const float*)d_in[11];
    const float* l0_bo = (const float*)d_in[12];
    const float* l0_Ws = (const float*)d_in[13];
    const float* l0_bs = (const float*)d_in[14];
    const float* l1_Wv = (const float*)d_in[15];
    const float* l1_Wk = (const float*)d_in[16];
    const float* l1_Wq = (const float*)d_in[17];
    const float* l1_Wo = (const float*)d_in[18];
    const float* l1_bo = (const float*)d_in[19];
    const float* l1_Ws = (const float*)d_in[20];
    const float* l1_bs = (const float*)d_in[21];
    const float* sc_W1 = (const float*)d_in[22];
    const float* sc_b1 = (const float*)d_in[23];
    const float* sc_W2 = (const float*)d_in[24];
    const float* sc_b2 = (const float*)d_in[25];

    const int E = in_sizes[0];
    const int N = in_sizes[3];

    // workspace layout
    char* w = (char*)d_ws;
    auto take = [&](size_t bytes) -> void* {
        void* p = (void*)w;
        w += (bytes + 255) & ~(size_t)255;
        return p;
    };
    float* logits = (float*)take((size_t)E * 4);
    int*   segmax = (int*)  take((size_t)N * 4);
    float* den    = (float*)take((size_t)N * 4);
    float* h0     = (float*)take((size_t)N * 2 * 4);
    float* h1     = (float*)take((size_t)N * HID * 4);
    float* h2     = (float*)take((size_t)N * HID * 4);
    float* agg    = (float*)take((size_t)N * HID * 4);
    float* hsum   = (float*)take(HID * 4);
    unsigned short* b_l0_Wv = (unsigned short*)take((size_t)64 * HID * 2);
    unsigned short* b_l0_Wk = (unsigned short*)take((size_t)64 * HID * 2);
    unsigned short* b_l0_Wq = (unsigned short*)take((size_t)32 * HID * 2);
    unsigned short* b_l0_Wo = (unsigned short*)take((size_t)64 * HID * 2);
    unsigned short* b_l0_Ws = (unsigned short*)take((size_t)32 * HID * 2);
    unsigned short* b_l1_Wv = (unsigned short*)take((size_t)96 * HID * 2);
    unsigned short* b_l1_Wk = (unsigned short*)take((size_t)64 * HID * 2);
    unsigned short* b_l1_Wq = (unsigned short*)take((size_t)64 * HID * 2);
    unsigned short* b_l1_Wo = (unsigned short*)take((size_t)64 * HID * 2);
    unsigned short* b_l1_Ws = (unsigned short*)take((size_t)64 * HID * 2);

    auto cvt = [&](const float* s, unsigned short* d, int Ksrc, int Kpad) {
        int tot = Kpad * HID;
        convert_w_swizzle<<<(tot + 255) / 256, 256, 0, stream>>>(s, d, Ksrc, Kpad);
    };
    cvt(l0_Wv, b_l0_Wv, 2 + TD, 64);
    cvt(l0_Wk, b_l0_Wk, 64, 64);
    cvt(l0_Wq, b_l0_Wq, 2, 32);
    cvt(l0_Wo, b_l0_Wo, 64, 64);
    cvt(l0_Ws, b_l0_Ws, 2, 32);
    cvt(l1_Wv, b_l1_Wv, 64 + TD, 96);
    cvt(l1_Wk, b_l1_Wk, 64, 64);
    cvt(l1_Wq, b_l1_Wq, 64, 64);
    cvt(l1_Wo, b_l1_Wo, 64, 64);
    cvt(l1_Ws, b_l1_Ws, 64, 64);

    build_h0<<<(N + 255) / 256, 256, 0, stream>>>(uf, vf, h0, N);

    const int eBlocks  = (E + 31) / 32;
    const int eThreads = (E + 255) / 256;
    const int nBlocks  = (N + 31) / 32;
    const int iBlocks  = (N * HID + 255) / 256;

    // ---- layer 0 (in_dim = 2) ----
    init_layer_ws<<<iBlocks, 256, 0, stream>>>(segmax, den, agg, N);
    tgat_logits<2, 64, 32><<<eBlocks, 256, 0, stream>>>(
        h0, dt, src, dst, tw, tb, b_l0_Wv, b_l0_Wk, b_l0_Wq, logits, segmax, E);
    tgat_expsum<<<eThreads, 256, 0, stream>>>(logits, dst, segmax, den, E);
    tgat_aggregate<2, 64><<<eBlocks, 256, 0, stream>>>(
        h0, dt, src, dst, tw, tb, b_l0_Wv, logits, den, agg, E);
    tgat_update<2, 32><<<nBlocks, 256, 0, stream>>>(
        h0, agg, b_l0_Wo, b_l0_Ws, l0_bo, l0_bs, h1, N);

    // ---- layer 1 (in_dim = 64) ----
    init_layer_ws<<<iBlocks, 256, 0, stream>>>(segmax, den, agg, N);
    tgat_logits<64, 96, 64><<<eBlocks, 256, 0, stream>>>(
        h1, dt, src, dst, tw, tb, b_l1_Wv, b_l1_Wk, b_l1_Wq, logits, segmax, E);
    tgat_expsum<<<eThreads, 256, 0, stream>>>(logits, dst, segmax, den, E);
    tgat_aggregate<64, 96><<<eBlocks, 256, 0, stream>>>(
        h1, dt, src, dst, tw, tb, b_l1_Wv, logits, den, agg, E);
    tgat_update<64, 64><<<nBlocks, 256, 0, stream>>>(
        h1, agg, b_l1_Wo, b_l1_Ws, l1_bo, l1_bs, h2, N);

    // ---- readout ----
    zero_vec<<<1, HID, 0, stream>>>(hsum, HID);
    reduce_sum<<<256, 256, 0, stream>>>(h2, hsum, N);
    scorer<<<1, HID, 0, stream>>>(hsum, sc_W1, sc_b1, sc_W2, sc_b2, (float*)d_out, N);
}
